// MambaBlock_65566970740818
// MI455X (gfx1250) — compile-verified, matrix operand order checked by reading, not run on hardware
//
#include <hip/hip_runtime.h>
#include <hip/hip_bf16.h>
#include <math.h>

// ---- problem constants ----
#define BATCH   2
#define SEQLEN  1024
#define D_MODEL 2048
#define D_INNER 4096
#define D_STATE 16
#define DT_RANK 128
#define XPROJ_N (DT_RANK + 2 * D_STATE)        // 160
#define MROWS   (BATCH * SEQLEN)               // 2048 token rows
#define PROJ_N  (2 * D_INNER)                  // 8192
#define OUT_OFF ((size_t)MROWS * D_MODEL)      // floats, offset of proj in d_out

typedef __attribute__((ext_vector_type(16))) __bf16 v16bf;
typedef __attribute__((ext_vector_type(8)))  __bf16 bf16x8;
typedef __attribute__((ext_vector_type(8)))  float  v8f;

// ---- tiled bf16 WMMA GEMM:  C[M,N] = A[M,K] * B[N,K]^T
// Block tile 128x64, K-step 64, double-buffered LDS, 8 waves each 32x32 (2x2 fragments).
#define BM 128
#define BN 64
#define BK 64
#define LDSS 72   // LDS row stride in bf16: 144 B = 36 banks -> 16 rows hit distinct banks

template<bool GUARD_N, bool EPI>
__global__ __launch_bounds__(256)
void gemm_wmma_bf16(const float* __restrict__ A, int lda,
                    const float* __restrict__ Bw, int ldb,
                    float* __restrict__ C, int ldc,
                    int Ndim, int Kdim,
                    const float* __restrict__ bias)
{
    __shared__ __bf16 As[2][BM * LDSS];
    __shared__ __bf16 Bs[2][BN * LDSS];

    const int tid  = threadIdx.x;
    const int lane = tid & 31;
    const int wave = tid >> 5;
    const int wm   = wave & 3;       // wave M position (4)
    const int wn   = wave >> 2;      // wave N position (2)
    const int g    = lane >> 4;      // K-half selector
    const int rr   = lane & 15;      // row within fragment

    const int blockN = blockIdx.x * BN;
    const int blockM = blockIdx.y * BM;

    v8f acc[2][2] = {};

    // tile-fill mapping: thread covers 8-float chunks; 32 rows per pass
    const int rA = tid >> 3;          // 0..31
    const int c8 = (tid & 7) << 3;    // 0,8,...,56

    float4 stA[4][2];                 // A: 4 passes x 8 floats
    float4 stB[2][2];                 // B: 2 passes x 8 floats

    auto loadRegs = [&](int k0) {
        #pragma unroll
        for (int p = 0; p < 4; ++p) {
            const float* src = A + (size_t)(blockM + p * 32 + rA) * lda + k0 + c8;
            stA[p][0] = *(const float4*)src;
            stA[p][1] = *(const float4*)(src + 4);
        }
        #pragma unroll
        for (int p = 0; p < 2; ++p) {
            const int n = blockN + p * 32 + rA;
            if (GUARD_N && n >= Ndim) {
                stB[p][0] = float4{0.f, 0.f, 0.f, 0.f};
                stB[p][1] = float4{0.f, 0.f, 0.f, 0.f};
            } else {
                const float* src = Bw + (size_t)n * ldb + k0 + c8;
                stB[p][0] = *(const float4*)src;
                stB[p][1] = *(const float4*)(src + 4);
            }
        }
    };

    auto storeLDS = [&](int buf) {
        #pragma unroll
        for (int p = 0; p < 4; ++p) {
            bf16x8 v;
            v[0] = (__bf16)stA[p][0].x; v[1] = (__bf16)stA[p][0].y;
            v[2] = (__bf16)stA[p][0].z; v[3] = (__bf16)stA[p][0].w;
            v[4] = (__bf16)stA[p][1].x; v[5] = (__bf16)stA[p][1].y;
            v[6] = (__bf16)stA[p][1].z; v[7] = (__bf16)stA[p][1].w;
            *(bf16x8*)&As[buf][(p * 32 + rA) * LDSS + c8] = v;
        }
        #pragma unroll
        for (int p = 0; p < 2; ++p) {
            bf16x8 v;
            v[0] = (__bf16)stB[p][0].x; v[1] = (__bf16)stB[p][0].y;
            v[2] = (__bf16)stB[p][0].z; v[3] = (__bf16)stB[p][0].w;
            v[4] = (__bf16)stB[p][1].x; v[5] = (__bf16)stB[p][1].y;
            v[6] = (__bf16)stB[p][1].z; v[7] = (__bf16)stB[p][1].w;
            *(bf16x8*)&Bs[buf][(p * 32 + rA) * LDSS + c8] = v;
        }
    };

    auto compute = [&](int buf) {
        #pragma unroll
        for (int ks = 0; ks < 2; ++ks) {           // two K=32 sub-steps
            v16bf af[2], bfr[2];
            #pragma unroll
            for (int i = 0; i < 2; ++i) {
                const __bf16* base = &As[buf][(wm * 32 + i * 16 + rr) * LDSS + ks * 32];
                bf16x8 lo = *(const bf16x8*)(base + 8 * g);
                bf16x8 hi = *(const bf16x8*)(base + 16 + 8 * g);
                #pragma unroll
                for (int e = 0; e < 8; ++e) { af[i][e] = lo[e]; af[i][e + 8] = hi[e]; }
            }
            #pragma unroll
            for (int j = 0; j < 2; ++j) {
                const __bf16* base = &Bs[buf][(wn * 32 + j * 16 + rr) * LDSS + ks * 32];
                bf16x8 lo = *(const bf16x8*)(base + 8 * g);
                bf16x8 hi = *(const bf16x8*)(base + 16 + 8 * g);
                #pragma unroll
                for (int e = 0; e < 8; ++e) { bfr[j][e] = lo[e]; bfr[j][e + 8] = hi[e]; }
            }
            #pragma unroll
            for (int i = 0; i < 2; ++i)
                #pragma unroll
                for (int j = 0; j < 2; ++j)
                    acc[i][j] = __builtin_amdgcn_wmma_f32_16x16x32_bf16(
                        false, af[i], false, bfr[j], (short)0, acc[i][j], false, false);
        }
    };

    // ---- software pipeline: prefetch next tile's globals during WMMA compute
    const int ntiles = Kdim / BK;
    loadRegs(0);
    storeLDS(0);
    __syncthreads();
    for (int kt = 0; kt < ntiles; ++kt) {
        if (kt + 1 < ntiles) loadRegs((kt + 1) * BK);   // issue loads, no wait yet
        compute(kt & 1);
        if (kt + 1 < ntiles) {
            storeLDS((kt + 1) & 1);                     // waits loadcnt here, after compute
            __syncthreads();
        }
    }

    // ---- store: element e of v8f -> row M = base + e + 8g, col N = base + rr
    #pragma unroll
    for (int j = 0; j < 2; ++j) {
        const int nIdx = blockN + wn * 32 + j * 16 + rr;
        if (GUARD_N && nIdx >= Ndim) continue;
        const float bv = EPI ? bias[nIdx] : 0.f;
        #pragma unroll
        for (int i = 0; i < 2; ++i) {
            #pragma unroll
            for (int e = 0; e < 8; ++e) {
                const int mIdx = blockM + wm * 32 + i * 16 + e + 8 * g;
                float v = acc[i][j][e];
                if (EPI) {                   // softplus(v + bias)
                    v += bv;
                    v = (v > 20.f) ? v : log1pf(__expf(v));
                }
                C[(size_t)mIdx * ldc + nIdx] = v;
            }
        }
    }
}

// ---- depthwise causal conv(4) + SiLU; proj (b,l,8192) channels [0,4096) -> xc (b,l,d)
__global__ __launch_bounds__(256)
void conv_silu_kernel(const float* __restrict__ proj,
                      const float* __restrict__ w,      // (4096,1,4)
                      const float* __restrict__ bias,
                      float* __restrict__ xc)
{
    const int idx = blockIdx.x * blockDim.x + threadIdx.x;  // b*L*E + l*E + d
    const int d = idx & (D_INNER - 1);
    const int l = (idx >> 12) & (SEQLEN - 1);
    const int b = idx >> 22;
    float acc = bias[d];
    const float* w4 = w + d * 4;
    #pragma unroll
    for (int k = 0; k < 4; ++k) {
        const int ll = l - 3 + k;
        if (ll >= 0)
            acc += w4[k] * proj[((size_t)(b * SEQLEN + ll)) * PROJ_N + d];
    }
    xc[idx] = acc / (1.f + __expf(-acc));   // SiLU
}

// ---- selective scan: one thread per channel d, h[16] in registers; y written over u in-place
__global__ __launch_bounds__(256)
void scan_kernel(const float* __restrict__ delta2,   // (b,l,d) post-softplus
                 const float* __restrict__ xdbl,     // (b,l,160): [128..144)=B, [144..160)=C
                 const float* __restrict__ proj,     // z at channel 4096+d
                 const float* __restrict__ A_log,    // (4096,16)
                 const float* __restrict__ Dp,       // (4096,)
                 float* __restrict__ xc)             // in: u, out: y*silu(z)
{
    const int b = blockIdx.x >> 4;
    const int d = ((blockIdx.x & 15) << 8) + threadIdx.x;

    float h[D_STATE], Av[D_STATE];
    #pragma unroll
    for (int n = 0; n < D_STATE; ++n) {
        h[n]  = 0.f;
        Av[n] = -__expf(A_log[d * D_STATE + n]);
    }
    const float Dd = Dp[d];

    for (int t = 0; t < SEQLEN; ++t) {
        const size_t m = (size_t)b * SEQLEN + t;
        const float dt = delta2[m * D_INNER + d];
        const float u  = xc[m * D_INNER + d];
        const float zv = proj[m * PROJ_N + D_INNER + d];
        const float* bc = xdbl + m * XPROJ_N + DT_RANK;   // uniform address -> broadcast
        const float du = dt * u;
        float y = 0.f;
        #pragma unroll
        for (int n = 0; n < D_STATE; ++n) {
            const float dA = __expf(dt * Av[n]);
            h[n] = dA * h[n] + du * bc[n];
            y += h[n] * bc[D_STATE + n];
        }
        const float yf = (y + u * Dd) * (zv / (1.f + __expf(-zv)));
        xc[m * D_INNER + d] = yf;
    }
}

extern "C" void kernel_launch(void* const* d_in, const int* in_sizes, int n_in,
                              void* d_out, int out_size, void* d_ws, size_t ws_size,
                              hipStream_t stream)
{
    const float* hidden = (const float*)d_in[1];
    const float* w_in   = (const float*)d_in[2];
    const float* conv_w = (const float*)d_in[3];
    const float* conv_b = (const float*)d_in[4];
    const float* xp_w   = (const float*)d_in[5];
    const float* dt_w   = (const float*)d_in[6];
    const float* dt_b   = (const float*)d_in[7];
    const float* out_w  = (const float*)d_in[8];
    const float* A_log  = (const float*)d_in[9];
    const float* Dp     = (const float*)d_in[10];

    float* out  = (float*)d_out;              // (2,1024,2048)
    float* proj = out + OUT_OFF;              // (2,1024,8192) — second tuple output

    float* ws     = (float*)d_ws;
    float* xc     = ws;                                   // (2048,4096) u, then y
    float* delta2 = ws + (size_t)MROWS * D_INNER;         // (2048,4096)
    float* xdbl   = delta2 + (size_t)MROWS * D_INNER;     // (2048,160)

    const dim3 blk(256);

    // 1) in_proj: proj = hidden @ W_in^T    (2048 x 8192 x 2048)
    gemm_wmma_bf16<false, false><<<dim3(PROJ_N / BN, MROWS / BM), blk, 0, stream>>>(
        hidden, D_MODEL, w_in, D_MODEL, proj, PROJ_N, PROJ_N, D_MODEL, nullptr);

    // 2) depthwise conv + SiLU -> xc (token-major)
    conv_silu_kernel<<<(BATCH * SEQLEN * D_INNER) / 256, blk, 0, stream>>>(
        proj, conv_w, conv_b, xc);

    // 3) x_proj: xdbl = xc @ Wx^T           (2048 x 160 x 4096), N edge guarded
    gemm_wmma_bf16<true, false><<<dim3((XPROJ_N + BN - 1) / BN, MROWS / BM), blk, 0, stream>>>(
        xc, D_INNER, xp_w, D_INNER, xdbl, XPROJ_N, XPROJ_N, D_INNER, nullptr);

    // 4) delta = softplus(dt @ Wdt^T + b)   (2048 x 4096 x 128); A rows are xdbl[:,0:128]
    gemm_wmma_bf16<false, true><<<dim3(D_INNER / BN, MROWS / BM), blk, 0, stream>>>(
        xdbl, XPROJ_N, dt_w, DT_RANK, delta2, D_INNER, D_INNER, DT_RANK, dt_b);

    // 5) selective scan (serial over L, parallel over 2*4096 channels); y overwrites u
    scan_kernel<<<BATCH * (D_INNER / 256), blk, 0, stream>>>(
        delta2, xdbl, proj, A_log, Dp, xc);

    // 6) out_proj: out = y @ Wout^T         (2048 x 2048 x 4096)
    gemm_wmma_bf16<false, false><<<dim3(D_MODEL / BN, MROWS / BM), blk, 0, stream>>>(
        xc, D_INNER, out_w, D_INNER, out, D_MODEL, D_MODEL, D_INNER, nullptr);
}
